// GPT_79328045957370
// MI455X (gfx1250) — matrix operand order repaired and checked by reading, not compiled, not run on hardware
//
#include <hip/hip_runtime.h>

#define B_ 4
#define T_ 1024
#define D_ 1024
#define H_ 16
#define HD_ 64
#define L_ 4
#define V_ 32000
#define EPS_ 1e-5f
#define IGNORE_ 12
#define ROWS 4096  // B_*T_

typedef __attribute__((ext_vector_type(16))) __bf16 bf16x16;
typedef __attribute__((ext_vector_type(8)))  float  floatx8;
typedef __attribute__((ext_vector_type(4)))  unsigned int v4u;
typedef __attribute__((ext_vector_type(8)))  unsigned int v8u;

union AFrag { bf16x16 v; unsigned int u[8]; };
union CAcc  { floatx8 v; float f[8]; };

__device__ __forceinline__ unsigned short f2bf(float x) {
  unsigned int u = __float_as_uint(x);
  u += 0x7FFFu + ((u >> 16) & 1u);   // round-to-nearest-even
  return (unsigned short)(u >> 16);
}

// --- CDNA5 TDM: DMA a [128 rows x 32 cols] bf16 tile into LDS.
// Descriptor: data_size=2B; pad 4 dwords (16B) after every 16 dwords (64B)
// -> LDS row stride = 32+8 = 40 elements, matching WMMA fragment loads.
__device__ __forceinline__ void tdm_load_tile_128x32(const unsigned short* gtile,
                                                     unsigned lds_addr, int stride_elems) {
  unsigned long long ga = (unsigned long long)gtile;
  v4u g0;
  g0[0] = 1u;                                   // count=1, user descriptor
  g0[1] = lds_addr;                             // LDS byte address
  g0[2] = (unsigned)ga;                         // global_addr[31:0]
  g0[3] = (unsigned)(ga >> 32) | (2u << 30);    // global_addr[56:32] | type=2
  v8u g1;
  g1[0] = (1u << 16) | (1u << 20) | (3u << 22) | (3u << 25); // data_size=2B, pad_en, int=64B, amt=16B
  g1[1] = ((unsigned)stride_elems & 0xFFFFu) << 16;          // tensor_dim0 lo16
  g1[2] = (((unsigned)stride_elems >> 16) & 0xFFFFu) | (0xFFFFu << 16); // dim0 hi | dim1 lo (big)
  g1[3] = (32u << 16);                          // tensor_dim1 hi=0 | tile_dim0=32
  g1[4] = 128u;                                 // tile_dim1=128 | tile_dim2=0
  g1[5] = (unsigned)stride_elems;               // tensor_dim0_stride lo32
  g1[6] = 0u;
  g1[7] = 0u;
  asm volatile("tensor_load_to_lds %0, %1" :: "s"(g0), "s"(g1) : "memory");
}

// --- CDNA5 async global->LDS copy (16B per lane), tracked by ASYNCcnt.
__device__ __forceinline__ void async_copy_b128(unsigned lds, unsigned long long gaddr) {
  asm volatile("global_load_async_to_lds_b128 %0, %1, off" :: "v"(lds), "v"(gaddr) : "memory");
}
__device__ __forceinline__ void wait_async0() {
  asm volatile("s_wait_asynccnt 0x0" ::: "memory");
}

// ---------------------------------------------------------------------------
// bf16 GEMM: C[M,N] = A[M,K] @ B[K,N] + bias (+resid)(relu)
// 256 thr = 8 waves; block tile 128x128; k-step 32; double-buffered LDS.
// A tile prefetched by TDM; B tile software-pipelined through VGPRs (transpose).
// ---------------------------------------------------------------------------
template<bool RES, bool RELU, bool OUTF, bool OUTB>
__global__ __launch_bounds__(256)
void gemm_bf16(const unsigned short* __restrict__ A,
               const unsigned short* __restrict__ Bm,
               const float* __restrict__ bias,
               const float* __restrict__ resid,
               float* __restrict__ Cf,
               unsigned short* __restrict__ Cb,
               int M, int N, int K)
{
  __shared__ __align__(16) unsigned short As[2][128 * 40];   // [row][k]+pad (TDM-padded)
  __shared__ __align__(16) unsigned short Bs[2][128 * 40];   // [n][k]+pad  (transposed)
  const int tid  = threadIdx.x;
  const int lane = tid & 31, wave = tid >> 5;
  const int lrow = lane & 15, half = lane >> 4;
  const int wm = wave & 3, wn = wave >> 2;
  const int m0 = blockIdx.y * 128, n0 = blockIdx.x * 128;
  const int bk = tid >> 3, bnb = (tid & 7) * 16;   // B stage: 8 thr per k-row, 16 bf16 each

  CAcc acc[2][4];
#pragma unroll
  for (int i = 0; i < 2; i++)
#pragma unroll
    for (int j = 0; j < 4; j++) acc[i][j].v = (floatx8)0.0f;

  auto storeB = [&](int buf, uint4 d0, uint4 d1) {
    unsigned int w[8] = {d0.x, d0.y, d0.z, d0.w, d1.x, d1.y, d1.z, d1.w};
#pragma unroll
    for (int j = 0; j < 8; j++) {
      unsigned int pair = w[j];
      Bs[buf][(bnb + 2 * j    ) * 40 + bk] = (unsigned short)(pair & 0xFFFFu);
      Bs[buf][(bnb + 2 * j + 1) * 40 + bk] = (unsigned short)(pair >> 16);
    }
  };

  // ---- prologue: tile 0 -> buffer 0
  if (wave == 0)
    tdm_load_tile_128x32(A + (size_t)m0 * K, (unsigned)(unsigned long long)&As[0][0], K);
  {
    const uint4* g = (const uint4*)(Bm + (size_t)bk * N + n0 + bnb);
    uint4 d0 = g[0], d1 = g[1];
    storeB(0, d0, d1);
  }
  if (wave == 0) __builtin_amdgcn_s_wait_tensorcnt(0);
  __syncthreads();

  const int nk = K / 32;
  for (int it = 0; it < nk; ++it) {
    const int c = it & 1;
    const bool hasNext = (it + 1 < nk);
    uint4 nb0, nb1;
    if (hasNext) {   // prefetch tile it+1 into buffer c^1
      int k0n = (it + 1) * 32;
      if (wave == 0)
        tdm_load_tile_128x32(A + (size_t)m0 * K + k0n,
                             (unsigned)(unsigned long long)&As[c ^ 1][0], K);
      const uint4* g = (const uint4*)(Bm + (size_t)(k0n + bk) * N + n0 + bnb);
      nb0 = g[0]; nb1 = g[1];
    }

    // ---- compute on buffer c
    AFrag af[2], bf[4];
#pragma unroll
    for (int mi = 0; mi < 2; mi++) {
      int row = wm * 32 + mi * 16 + lrow;
#pragma unroll
      for (int v = 0; v < 8; v++) {
        int kidx = (v & 3) * 2 + (v >> 2) * 16 + half * 8;
        af[mi].u[v] = *(const unsigned int*)(&As[c][row * 40 + kidx]);
      }
    }
#pragma unroll
    for (int ni = 0; ni < 4; ni++) {
      int col = wn * 64 + ni * 16 + lrow;
#pragma unroll
      for (int v = 0; v < 8; v++)
        bf[ni].u[v] = *(const unsigned int*)(&Bs[c][col * 40 + (v & 3) * 2 + (v >> 2) * 16 + half * 8]);
    }
#pragma unroll
    for (int mi = 0; mi < 2; mi++)
#pragma unroll
      for (int ni = 0; ni < 4; ni++)
        acc[mi][ni].v = __builtin_amdgcn_wmma_f32_16x16x32_bf16(
            false, af[mi].v, false, bf[ni].v, (short)0, acc[mi][ni].v, false, false);

    if (hasNext) storeB(c ^ 1, nb0, nb1);
    if (wave == 0 && hasNext) __builtin_amdgcn_s_wait_tensorcnt(0);
    __syncthreads();
  }

  // ---- epilogue (fully specialized, no runtime branches)
#pragma unroll
  for (int mi = 0; mi < 2; mi++)
#pragma unroll
    for (int ni = 0; ni < 4; ni++) {
      int col = n0 + wn * 64 + ni * 16 + lrow;
      float bv = bias[col];
#pragma unroll
      for (int e = 0; e < 8; e++) {
        int row = m0 + wm * 32 + mi * 16 + e + 8 * half;
        size_t idx = (size_t)row * N + col;
        float val = acc[mi][ni].f[e] + bv;
        if constexpr (RES)  val += resid[idx];
        if constexpr (RELU) val = val > 0.0f ? val : 0.0f;
        if constexpr (OUTF) Cf[idx] = val;
        if constexpr (OUTB) Cb[idx] = f2bf(val);
      }
    }
}

// ---------------------------------------------------------------------------
// Flash attention: one block per (b,h, 64 q rows); 128 thr = 4 waves.
// k tile loaded via async global->LDS (no transpose needed); v transposed manually.
// ---------------------------------------------------------------------------
__global__ __launch_bounds__(128)
void attn_kernel(const unsigned short* __restrict__ q,
                 const unsigned short* __restrict__ k,
                 const unsigned short* __restrict__ v,
                 unsigned short* __restrict__ y, float scale)
{
  __shared__ __align__(16) unsigned short k_lds[64 * 72];   // [key][d]
  __shared__ __align__(16) unsigned short vT_lds[64 * 72];  // [d][key]
  __shared__ __align__(16) unsigned short p_lds[64 * 72];   // [qrow][key] (wave-private slices)

  const int tid = threadIdx.x;
  const int lane = tid & 31, wave = tid >> 5;
  const int lrow = lane & 15, half = lane >> 4;
  const int qb = blockIdx.x, bh = blockIdx.y;
  const int b = bh / H_, hh = bh % H_;
  const size_t base = (size_t)bh * T_ * HD_;

  AFrag qa[2];   // q rows (A frags), resident for whole kernel
  {
    int trow = qb * 64 + wave * 16 + lrow;
    const unsigned short* qrow = q + base + (size_t)trow * HD_;
#pragma unroll
    for (int f = 0; f < 2; f++)
#pragma unroll
      for (int vv = 0; vv < 8; vv++) {
        int kidx = f * 32 + (vv & 3) * 2 + (vv >> 2) * 16 + half * 8;
        qa[f].u[vv] = *(const unsigned int*)(qrow + kidx);
      }
  }

  CAcc o[4];
#pragma unroll
  for (int ni = 0; ni < 4; ni++) o[ni].v = (floatx8)0.0f;
  float m8[8], l8[8];
#pragma unroll
  for (int e = 0; e < 8; e++) { m8[e] = -__builtin_inff(); l8[e] = 0.0f; }

  const int ldk = tid >> 1, ldc = (tid & 1) * 32;

  for (int kb = 0; kb <= qb; kb++) {
    { // k tile: async copy straight into LDS (natural layout)
      unsigned klds = (unsigned)(unsigned long long)&k_lds[ldk * 72 + ldc];
      unsigned long long kg =
          (unsigned long long)(k + base + (size_t)(kb * 64 + ldk) * HD_ + ldc);
      async_copy_b128(klds,       kg);
      async_copy_b128(klds + 16u, kg + 16ull);
      async_copy_b128(klds + 32u, kg + 32ull);
      async_copy_b128(klds + 48u, kg + 48ull);
    }
    { // v tile: manual transpose into LDS
      const uint4* gv = (const uint4*)(v + base + (size_t)(kb * 64 + ldk) * HD_ + ldc);
      uint4 d[4] = {gv[0], gv[1], gv[2], gv[3]};
      const unsigned int* w = (const unsigned int*)d;
#pragma unroll
      for (int j = 0; j < 16; j++) {
        unsigned int pair = w[j];
        vT_lds[(ldc + 2 * j    ) * 72 + ldk] = (unsigned short)(pair & 0xFFFFu);
        vT_lds[(ldc + 2 * j + 1) * 72 + ldk] = (unsigned short)(pair >> 16);
      }
    }
    wait_async0();
    __syncthreads();

    // S = q @ k^T  (16 q rows x 64 keys per wave)
    CAcc s[4];
#pragma unroll
    for (int ni = 0; ni < 4; ni++) s[ni].v = (floatx8)0.0f;
#pragma unroll
    for (int f = 0; f < 2; f++) {
      AFrag kf[4];
#pragma unroll
      for (int ni = 0; ni < 4; ni++) {
        int key = ni * 16 + lrow;
#pragma unroll
        for (int vv = 0; vv < 8; vv++)
          kf[ni].u[vv] = *(const unsigned int*)(&k_lds[key * 72 + f * 32 + (vv & 3) * 2 + (vv >> 2) * 16 + half * 8]);
      }
#pragma unroll
      for (int ni = 0; ni < 4; ni++)
        s[ni].v = __builtin_amdgcn_wmma_f32_16x16x32_bf16(
            false, qa[f].v, false, kf[ni].v, (short)0, s[ni].v, false, false);
    }

    // scale + causal mask
#pragma unroll
    for (int ni = 0; ni < 4; ni++) {
      int col = kb * 64 + ni * 16 + lrow;
#pragma unroll
      for (int e = 0; e < 8; e++) {
        float sv = s[ni].f[e] * scale;
        int row = qb * 64 + wave * 16 + e + 8 * half;
        if (kb == qb && col > row) sv = -__builtin_inff();
        s[ni].f[e] = sv;
      }
    }

    // online softmax: row max across the 16 lanes of this half
    float mnew[8];
#pragma unroll
    for (int e = 0; e < 8; e++) {
      float mx = fmaxf(fmaxf(s[0].f[e], s[1].f[e]), fmaxf(s[2].f[e], s[3].f[e]));
#pragma unroll
      for (int msk = 1; msk < 16; msk <<= 1) mx = fmaxf(mx, __shfl_xor(mx, msk, 32));
      mnew[e] = fmaxf(m8[e], mx);
    }
    float rs[8];
#pragma unroll
    for (int e = 0; e < 8; e++) rs[e] = 0.0f;
#pragma unroll
    for (int ni = 0; ni < 4; ni++)
#pragma unroll
      for (int e = 0; e < 8; e++) {
        float p = expf(s[ni].f[e] - mnew[e]);
        rs[e] += p;
        p_lds[(wave * 16 + e + 8 * half) * 72 + ni * 16 + lrow] = f2bf(p);
      }
#pragma unroll
    for (int e = 0; e < 8; e++) {
      float r = rs[e];
#pragma unroll
      for (int msk = 1; msk < 16; msk <<= 1) r += __shfl_xor(r, msk, 32);
      float alpha = expf(m8[e] - mnew[e]);
      l8[e] = l8[e] * alpha + r;
      m8[e] = mnew[e];
#pragma unroll
      for (int ni = 0; ni < 4; ni++) o[ni].f[e] *= alpha;
    }

    // o += p @ v
#pragma unroll
    for (int f = 0; f < 2; f++) {
      AFrag pa;
      {
        int prow = wave * 16 + lrow;
#pragma unroll
        for (int vv = 0; vv < 8; vv++) {
          int kidx = f * 32 + (vv & 3) * 2 + (vv >> 2) * 16 + half * 8;
          pa.u[vv] = *(const unsigned int*)(&p_lds[prow * 72 + kidx]);
        }
      }
#pragma unroll
      for (int ni = 0; ni < 4; ni++) {
        AFrag vb;
        int col = ni * 16 + lrow;
#pragma unroll
        for (int vv = 0; vv < 8; vv++)
          vb.u[vv] = *(const unsigned int*)(&vT_lds[col * 72 + f * 32 + (vv & 3) * 2 + (vv >> 2) * 16 + half * 8]);
        o[ni].v = __builtin_amdgcn_wmma_f32_16x16x32_bf16(
            false, pa.v, false, vb.v, (short)0, o[ni].v, false, false);
      }
    }
    __syncthreads();
  }

  // normalize and write into [B,T,D]
#pragma unroll
  for (int ni = 0; ni < 4; ni++) {
    int col = hh * HD_ + ni * 16 + lrow;
#pragma unroll
    for (int e = 0; e < 8; e++) {
      int trow = qb * 64 + wave * 16 + e + 8 * half;
      y[(size_t)(b * T_ + trow) * D_ + col] = f2bf(o[ni].f[e] / l8[e]);
    }
  }
}

// ---------------------------------------------------------------------------
// Small kernels
// ---------------------------------------------------------------------------
__global__ void embed_kernel(const int* __restrict__ x, const float* __restrict__ tok,
                             const float* __restrict__ pos, float* __restrict__ h)
{
  int row = blockIdx.x, t = row % T_;
  int id = x[row];
  const float* tr = tok + (size_t)id * D_;
  const float* pr = pos + (size_t)t * D_;
  float* hr = h + (size_t)row * D_;
  for (int i = threadIdx.x; i < D_; i += blockDim.x) hr[i] = tr[i] + pr[i];
}

__global__ __launch_bounds__(256)
void layernorm_bf16(const float* __restrict__ x, const float* __restrict__ w,
                    const float* __restrict__ bvec, unsigned short* __restrict__ out)
{
  __shared__ float red[256];
  const int row = blockIdx.x, tid = threadIdx.x;
  const float* xr = x + (size_t)row * D_;
  float s = 0.f;
  for (int i = tid; i < D_; i += 256) s += xr[i];
  red[tid] = s; __syncthreads();
  for (int st = 128; st > 0; st >>= 1) { if (tid < st) red[tid] += red[tid + st]; __syncthreads(); }
  float mean = red[0] / D_;
  __syncthreads();
  float s2 = 0.f;
  for (int i = tid; i < D_; i += 256) { float d = xr[i] - mean; s2 += d * d; }
  red[tid] = s2; __syncthreads();
  for (int st = 128; st > 0; st >>= 1) { if (tid < st) red[tid] += red[tid + st]; __syncthreads(); }
  float rstd = rsqrtf(red[0] / D_ + EPS_);
  for (int i = tid; i < D_; i += 256)
    out[(size_t)row * D_ + i] = f2bf((xr[i] - mean) * rstd * w[i] + bvec[i]);
}

__global__ void qkv_split(const unsigned short* __restrict__ qkv,
                          unsigned short* __restrict__ q, unsigned short* __restrict__ k,
                          unsigned short* __restrict__ v)
{
  int row = blockIdx.x, b = row / T_, t = row % T_;
  const unsigned short* src = qkv + (size_t)row * (3 * D_);
  for (int c = threadIdx.x; c < D_; c += blockDim.x) {
    int head = c >> 6, d = c & 63;
    size_t dst = ((size_t)(b * H_ + head) * T_ + t) * HD_ + d;
    q[dst] = src[c];
    k[dst] = src[c + D_];
    v[dst] = src[c + 2 * D_];
  }
}

__global__ void f32_to_bf16(const float* __restrict__ in, unsigned short* __restrict__ out, size_t n)
{
  size_t i = (size_t)blockIdx.x * blockDim.x + threadIdx.x;
  size_t stride = (size_t)gridDim.x * blockDim.x;
  for (; i < n; i += stride) out[i] = f2bf(in[i]);
}

__global__ void ce_init(float* acc) { acc[0] = 0.f; acc[1] = 0.f; }

__global__ __launch_bounds__(256)
void ce_rows(const float* __restrict__ logits, const int* __restrict__ targets, float* acc)
{
  __shared__ float red[256];
  int row = blockIdx.x, tid = threadIdx.x;
  const float* lr = logits + (size_t)row * V_;
  float mx = -__builtin_inff();
  for (int i = tid; i < V_; i += 256) mx = fmaxf(mx, lr[i]);
  red[tid] = mx; __syncthreads();
  for (int st = 128; st > 0; st >>= 1) { if (tid < st) red[tid] = fmaxf(red[tid], red[tid + st]); __syncthreads(); }
  mx = red[0]; __syncthreads();
  float s = 0.f;
  for (int i = tid; i < V_; i += 256) s += expf(lr[i] - mx);
  red[tid] = s; __syncthreads();
  for (int st = 128; st > 0; st >>= 1) { if (tid < st) red[tid] += red[tid + st]; __syncthreads(); }
  if (tid == 0) {
    int tgt = targets[row];
    if (tgt != IGNORE_) {
      float lp = lr[tgt] - mx - logf(red[0]);
      atomicAdd(&acc[0], -lp);
      atomicAdd(&acc[1], 1.0f);
    }
  }
}

__global__ void ce_final(const float* acc, float* out) { out[0] = acc[0] / fmaxf(acc[1], 1.0f); }

// ---------------------------------------------------------------------------
extern "C" void kernel_launch(void* const* d_in, const int* in_sizes, int n_in,
                              void* d_out, int out_size, void* d_ws, size_t ws_size,
                              hipStream_t stream)
{
  (void)in_sizes; (void)n_in; (void)out_size; (void)ws_size;
  const int*   x    = (const int*)  d_in[0];
  const int*   tgt  = (const int*)  d_in[1];
  const float* tok  = (const float*)d_in[2];
  const float* pos  = (const float*)d_in[3];
  const float* ln1w = (const float*)d_in[4];
  const float* ln1b = (const float*)d_in[5];
  const float* ln2w = (const float*)d_in[6];
  const float* ln2b = (const float*)d_in[7];
  const float* qkvw = (const float*)d_in[8];
  const float* qkvb = (const float*)d_in[9];
  const float* projw= (const float*)d_in[10];
  const float* projb= (const float*)d_in[11];
  const float* ff1w = (const float*)d_in[12];
  const float* ff1b = (const float*)d_in[13];
  const float* ff2w = (const float*)d_in[14];
  const float* ff2b = (const float*)d_in[15];
  const float* lmw  = (const float*)d_in[16];
  const float* lmb  = (const float*)d_in[17];

  char* ws = (char*)d_ws;
  size_t off = 0;
  auto alloc = [&](size_t bytes) -> void* {
    void* p = ws + off;
    off += (bytes + 255) & ~(size_t)255;
    return p;
  };
  float*          h      = (float*)         alloc((size_t)ROWS * D_ * 4);
  unsigned short* a_bf   = (unsigned short*)alloc((size_t)ROWS * D_ * 2);
  unsigned short* qkv_bf = (unsigned short*)alloc((size_t)ROWS * 3 * D_ * 2);
  unsigned short* q_bf   = (unsigned short*)alloc((size_t)ROWS * D_ * 2);
  unsigned short* k_bf   = (unsigned short*)alloc((size_t)ROWS * D_ * 2);
  unsigned short* v_bf   = (unsigned short*)alloc((size_t)ROWS * D_ * 2);
  unsigned short* y_bf   = (unsigned short*)alloc((size_t)ROWS * D_ * 2);
  unsigned short* ff_bf  = (unsigned short*)alloc((size_t)ROWS * 4 * D_ * 2);
  unsigned short* wbuf   = (unsigned short*)alloc((size_t)D_ * V_ * 2);
  float*          acc    = (float*)         alloc(256);

  float* logits = (float*)d_out;
  float* loss   = logits + (size_t)ROWS * V_;
  const float scale = 0.03125f;  // 1/sqrt(D): reference scales by n_embd, not head_dim

  embed_kernel<<<ROWS, 256, 0, stream>>>(x, tok, pos, h);

  for (int l = 0; l < L_; l++) {
    // ---- attention half ----
    layernorm_bf16<<<ROWS, 256, 0, stream>>>(h, ln1w + l * D_, ln1b + l * D_, a_bf);
    f32_to_bf16<<<4096, 256, 0, stream>>>(qkvw + (size_t)l * D_ * 3 * D_, wbuf, (size_t)D_ * 3 * D_);
    gemm_bf16<false, false, false, true><<<dim3(3 * D_ / 128, ROWS / 128), 256, 0, stream>>>(
        a_bf, wbuf, qkvb + l * 3 * D_, nullptr, nullptr, qkv_bf, ROWS, 3 * D_, D_);
    qkv_split<<<ROWS, 256, 0, stream>>>(qkv_bf, q_bf, k_bf, v_bf);
    attn_kernel<<<dim3(T_ / 64, B_ * H_), 128, 0, stream>>>(q_bf, k_bf, v_bf, y_bf, scale);
    f32_to_bf16<<<4096, 256, 0, stream>>>(projw + (size_t)l * D_ * D_, wbuf, (size_t)D_ * D_);
    gemm_bf16<true, false, true, false><<<dim3(D_ / 128, ROWS / 128), 256, 0, stream>>>(
        y_bf, wbuf, projb + l * D_, h, h, nullptr, ROWS, D_, D_);
    // ---- MLP half ----
    layernorm_bf16<<<ROWS, 256, 0, stream>>>(h, ln2w + l * D_, ln2b + l * D_, a_bf);
    f32_to_bf16<<<4096, 256, 0, stream>>>(ff1w + (size_t)l * D_ * 4 * D_, wbuf, (size_t)D_ * 4 * D_);
    gemm_bf16<false, true, false, true><<<dim3(4 * D_ / 128, ROWS / 128), 256, 0, stream>>>(
        a_bf, wbuf, ff1b + l * 4 * D_, nullptr, nullptr, ff_bf, ROWS, 4 * D_, D_);
    f32_to_bf16<<<4096, 256, 0, stream>>>(ff2w + (size_t)l * 4 * D_ * D_, wbuf, (size_t)4 * D_ * D_);
    gemm_bf16<true, false, true, false><<<dim3(D_ / 128, ROWS / 128), 256, 0, stream>>>(
        ff_bf, wbuf, ff2b + l * D_, h, h, nullptr, ROWS, D_, 4 * D_);
  }

  // LM head: logits = h @ lm_w + lm_b  (32000 = 250 * 128)
  f32_to_bf16<<<4096, 256, 0, stream>>>(h, a_bf, (size_t)ROWS * D_);
  f32_to_bf16<<<4096, 256, 0, stream>>>(lmw, wbuf, (size_t)D_ * V_);
  gemm_bf16<false, false, true, false><<<dim3(V_ / 128, ROWS / 128), 256, 0, stream>>>(
      a_bf, wbuf, lmb, nullptr, logits, nullptr, ROWS, V_, D_);

  // cross-entropy (mean over targets != IGNORE)
  ce_init<<<1, 1, 0, stream>>>(acc);
  ce_rows<<<ROWS, 256, 0, stream>>>(logits, tgt, acc);
  ce_final<<<1, 1, 0, stream>>>(acc, loss);
}
